// AttentionBlock_49065706390163
// MI455X (gfx1250) — matrix operand order repaired and checked by reading, not compiled, not run on hardware
//
#include <hip/hip_runtime.h>
#include <hip/hip_bf16.h>
#include <stdint.h>

typedef _Float16 h16;
typedef __attribute__((ext_vector_type(16))) _Float16 v16h;
typedef __attribute__((ext_vector_type(8)))  _Float16 v8h;
typedef __attribute__((ext_vector_type(8)))  float    v8f;

union V16H { v16h v; v8h h[2]; uint32_t u[8]; };

// ---------------------------------------------------------------------------
// WMMA helper: D = A(16x32 f16) * B(32x16 f16) + C(16x16 f32)
// ---------------------------------------------------------------------------
__device__ __forceinline__ v8f wmma16(v16h a, v16h b, v8f c) {
  return __builtin_amdgcn_wmma_f32_16x16x32_f16(
      /*neg_a=*/false, a, /*neg_b=*/false, b,
      /*c_mod=*/(short)0, c, /*reuse_a=*/false, /*reuse_b=*/false);
}

// A-fragment (16 rows x 32-K chunk), row-major source with given row stride.
// Lane l: row = l%16; halves e -> K = 16*hs+e
__device__ __forceinline__ v16h load_afrag(const h16* rowbase, int stride, int lane) {
  int m = lane & 15, hs = lane >> 4;
  const h16* r = rowbase + (size_t)m * stride;
  V16H t;
  t.h[0] = *(const v8h*)(r + 16 * hs);
  t.h[1] = *(const v8h*)(r + 16 * hs + 8);
  return t.v;
}

// B-fragment (32-K chunk x 16 cols) where column n of B is row n of the
// row-major source (i.e. B = src^T). Lane l: col = l%16; halves e -> K=16*hs+e
__device__ __forceinline__ v16h load_bfrag(const h16* colbase, int stride, int lane) {
  int n = lane & 15, hs = lane >> 4;
  const h16* r = colbase + (size_t)n * stride + 16 * hs;
  V16H t;
  t.h[0] = *(const v8h*)(r);
  t.h[1] = *(const v8h*)(r + 8);
  return t.v;
}

__device__ __forceinline__ uint32_t pkh(float a, float b) {
  union { _Float16 h[2]; uint32_t u; } t;
  t.h[0] = (_Float16)a; t.h[1] = (_Float16)b;
  return t.u;
}

// CDNA5 async copy: 16 bytes global -> LDS, tracked by ASYNCcnt (no VGPR data).
__device__ __forceinline__ void async_b128(uint32_t lds_off, const h16* gptr) {
  asm volatile("global_load_async_to_lds_b128 %0, %1, off"
               :: "v"(lds_off), "v"((uint64_t)(uintptr_t)gptr)
               : "memory");
}
__device__ __forceinline__ void wait_async8() {
  asm volatile("s_wait_asynccnt 8" ::: "memory");
}
__device__ __forceinline__ void wait_async0() {
  asm volatile("s_wait_asynccnt 0" ::: "memory");
}

// ---------------------------------------------------------------------------
// Kernel 1: GroupNorm statistics. One block per (batch,group): 2 ch x 4096.
// (b*32+g)*8192 == ((b*64 + 2g)*4096), so input is contiguous per block.
// ---------------------------------------------------------------------------
__global__ void __launch_bounds__(256)
gn_stats_kernel(const float* __restrict__ x, float* __restrict__ stats) {
  int bg = blockIdx.x;                       // 0..127
  const float* p = x + (size_t)bg * 8192;
  float s = 0.f, ss = 0.f;
  for (int i = threadIdx.x; i < 8192; i += 256) {
    float v = p[i];
    s += v;
    ss = fmaf(v, v, ss);
  }
  #pragma unroll
  for (int m = 16; m >= 1; m >>= 1) {
    s  += __shfl_xor(s,  m, 32);
    ss += __shfl_xor(ss, m, 32);
  }
  __shared__ float red[16];
  int wid = threadIdx.x >> 5;
  if ((threadIdx.x & 31) == 0) { red[wid * 2] = s; red[wid * 2 + 1] = ss; }
  __syncthreads();
  if (threadIdx.x == 0) {
    float S = 0.f, SS = 0.f;
    #pragma unroll
    for (int w = 0; w < 8; ++w) { S += red[w * 2]; SS += red[w * 2 + 1]; }
    float mean = S * (1.f / 8192.f);
    float var  = SS * (1.f / 8192.f) - mean * mean;
    stats[bg * 2]     = mean;
    stats[bg * 2 + 1] = rsqrtf(var + 1e-5f);
  }
}

// ---------------------------------------------------------------------------
// Kernel 2: GN apply + fused QKV 1x1 convs.
// Block = 64 spatial positions of one batch. Writes:
//   q[b][pos][c] (f16, pre-scaled by 1/8), k[b][pos][c] (f16), vT[b][c][pos].
// ---------------------------------------------------------------------------
__global__ void __launch_bounds__(256)
qkv_kernel(const float* __restrict__ x, const float* __restrict__ stats,
           const float* __restrict__ Wq, const float* __restrict__ bq,
           const float* __restrict__ Wk, const float* __restrict__ bk,
           const float* __restrict__ Wv, const float* __restrict__ bv,
           const float* __restrict__ gamma, const float* __restrict__ beta,
           h16* __restrict__ q, h16* __restrict__ k, h16* __restrict__ vT) {
  int blk = blockIdx.x;                 // 0..255
  int b = blk >> 6;
  int p0 = (blk & 63) * 64;
  __shared__ float xn[64][65];
  __shared__ float wq_s[4096], wk_s[4096], wv_s[4096];
  __shared__ float bias[192];
  int tid = threadIdx.x;
  for (int i = tid; i < 4096; i += 256) {
    wq_s[i] = Wq[i]; wk_s[i] = Wk[i]; wv_s[i] = Wv[i];
    int ch = i >> 6, pos = i & 63;
    int sg = (b * 32 + (ch >> 1)) * 2;
    float mean = stats[sg], rstd = stats[sg + 1];
    float v = x[((size_t)b * 64 + ch) * 4096 + p0 + pos];
    xn[pos][ch] = (v - mean) * rstd * gamma[ch] + beta[ch];
  }
  if (tid < 192)
    bias[tid] = (tid < 64) ? bq[tid] : (tid < 128 ? bk[tid - 64] : bv[tid - 128]);
  __syncthreads();

  int pos = tid >> 2;
  int oc0 = (tid & 3) * 16;
  const float* xr = xn[pos];
  for (int oc = oc0; oc < oc0 + 16; ++oc) {
    float dq = 0.f, dk = 0.f, dv = 0.f;
    const float* wqr = &wq_s[oc * 64];
    const float* wkr = &wk_s[oc * 64];
    const float* wvr = &wv_s[oc * 64];
    #pragma unroll
    for (int c = 0; c < 64; ++c) {
      float xv = xr[c];
      dq = fmaf(xv, wqr[c], dq);
      dk = fmaf(xv, wkr[c], dk);
      dv = fmaf(xv, wvr[c], dv);
    }
    size_t pg = (size_t)b * 4096 + p0 + pos;
    q[pg * 64 + oc] = (h16)((dq + bias[oc]) * 0.125f);   // 1/sqrt(64) folded in
    k[pg * 64 + oc] = (h16)(dk + bias[64 + oc]);
    vT[((size_t)b * 64 + oc) * 4096 + p0 + pos] = (h16)(dv + bias[128 + oc]);
  }
}

// ---------------------------------------------------------------------------
// Kernel 3: flash attention. One wave32 owns 16 queries, streams all 4096
// keys in blocks of 64. S^T = K * Q^T so the softmax row reduction is
// per-lane + one shfl_xor(16); the packed-f16 P^T D-registers map onto the
// second GEMM's B-fragment with 4 shfl_xor(16) per 32-key chunk.
// K/V tiles (8 KB each) are staged into LDS once per block with CDNA5
// global_load_async_to_lds_b128 (double-buffered, ASYNCcnt-pipelined); all
// 8 waves of the block then read fragments via ds_load_b128. Rows padded to
// 144 B so the 16-lane strided b128 fragment reads are bank-conflict-free.
// Output O^T written channel-major: ao[b][c][n] (f32).
// ---------------------------------------------------------------------------
#define LROW 72   // 64 halves payload + 8 halves pad (144 B row stride)

__global__ void __launch_bounds__(256)
flash_kernel(const h16* __restrict__ q, const h16* __restrict__ k,
             const h16* __restrict__ vT, float* __restrict__ ao) {
  __shared__ h16 ksh[2][64][LROW];   // 18 KB
  __shared__ h16 vsh[2][64][LROW];   // 18 KB

  int tid  = threadIdx.x;
  int lane = tid & 31;
  int wave = tid >> 5;
  int hs   = lane >> 4;
  int qcol = lane & 15;
  int b = blockIdx.y;
  int qbase = (blockIdx.x * 8 + wave) * 16;

  const h16* qb = q  + (size_t)b * 4096 * 64;
  const h16* kb = k  + (size_t)b * 4096 * 64;
  const h16* vb = vT + (size_t)b * 64 * 4096;

  // Q^T B-fragments for this wave's 16 queries (resident for whole loop)
  v16h qf0 = load_bfrag(qb + (size_t)qbase * 64 + 0,  64, lane);
  v16h qf1 = load_bfrag(qb + (size_t)qbase * 64 + 32, 64, lane);

  // stage one 64-key tile (K: [key][ch], V: [ch][key]) -> 4 async b128/thread
  auto stage = [&](int buf, int kb0) {
    #pragma unroll
    for (int r = 0; r < 2; ++r) {
      int i = tid + r * 256;             // 0..511
      int row = i >> 3, seg = (i & 7) * 8;
      async_b128((uint32_t)(uintptr_t)&ksh[buf][row][seg],
                 kb + (size_t)(kb0 + row) * 64 + seg);
      async_b128((uint32_t)(uintptr_t)&vsh[buf][row][seg],
                 vb + (size_t)row * 4096 + kb0 + seg);
    }
  };

  float m_run = -3.0e38f, l_run = 0.f;
  v8f acc[4];
  #pragma unroll
  for (int mt = 0; mt < 4; ++mt) acc[mt] = (v8f){};

  stage(0, 0);

  for (int it = 0; it < 64; ++it) {
    int buf = it & 1;
    int kb0 = it * 64;
    if (it + 1 < 64) {
      stage(buf ^ 1, kb0 + 64);
      wait_async8();                 // in-order: tile `it` is the oldest 4 of 8 issued
    } else {
      wait_async0();
    }
    __syncthreads();                 // tile `it` visible to all waves

    // ---- S^T tiles: 4 key-tiles x (K=64 in 2 chunks) = 8 WMMAs (from LDS)
    v8f s[4];
    #pragma unroll
    for (int t = 0; t < 4; ++t) {
      v8f c = (v8f){};
      c = wmma16(load_afrag(&ksh[buf][16 * t][0],  LROW, lane), qf0, c);
      c = wmma16(load_afrag(&ksh[buf][16 * t][32], LROW, lane), qf1, c);
      s[t] = c;
    }

    // ---- online softmax (pairwise trees to shorten dependency chains)
    float tm[4];
    #pragma unroll
    for (int t = 0; t < 4; ++t) {
      float a = fmaxf(fmaxf(s[t][0], s[t][1]), fmaxf(s[t][2], s[t][3]));
      float c = fmaxf(fmaxf(s[t][4], s[t][5]), fmaxf(s[t][6], s[t][7]));
      tm[t] = fmaxf(a, c);
    }
    float bm = fmaxf(fmaxf(tm[0], tm[1]), fmaxf(tm[2], tm[3]));
    bm = fmaxf(bm, __shfl_xor(bm, 16, 32));
    float m_new = fmaxf(m_run, bm);
    float alpha = __expf(m_run - m_new);

    uint32_t p[4][4];
    float ps[4];
    #pragma unroll
    for (int t = 0; t < 4; ++t) {
      float e[8];
      #pragma unroll
      for (int r = 0; r < 8; ++r) e[r] = __expf(s[t][r] - m_new);
      #pragma unroll
      for (int i = 0; i < 4; ++i) p[t][i] = pkh(e[2 * i], e[2 * i + 1]);
      ps[t] = ((e[0] + e[1]) + (e[2] + e[3])) + ((e[4] + e[5]) + (e[6] + e[7]));
    }
    float bsum = (ps[0] + ps[1]) + (ps[2] + ps[3]);
    bsum += __shfl_xor(bsum, 16, 32);
    l_run = l_run * alpha + bsum;
    m_run = m_new;

    #pragma unroll
    for (int mt = 0; mt < 4; ++mt)
      #pragma unroll
      for (int r = 0; r < 8; ++r) acc[mt][r] *= alpha;

    // ---- O^T += V^T * P^T : build P^T B-fragments from D regs (half swap)
    #pragma unroll
    for (int c2 = 0; c2 < 2; ++c2) {
      V16H pb;
      #pragma unroll
      for (int i = 0; i < 4; ++i) {
        uint32_t op = hs ? p[2 * c2][i] : p[2 * c2 + 1][i];
        uint32_t z = __shfl_xor(op, 16, 32);
        pb.u[i]     = hs ? z : p[2 * c2][i];
        pb.u[4 + i] = hs ? p[2 * c2 + 1][i] : z;
      }
      #pragma unroll
      for (int mt = 0; mt < 4; ++mt) {
        v16h va = load_afrag(&vsh[buf][16 * mt][32 * c2], LROW, lane);
        acc[mt] = wmma16(va, pb.v, acc[mt]);
      }
    }

    __syncthreads();                 // all waves done with `buf` before reuse
  }

  float inv = 1.0f / l_run;
  int qg = qbase + qcol;
  #pragma unroll
  for (int mt = 0; mt < 4; ++mt)
    #pragma unroll
    for (int r = 0; r < 8; ++r) {
      int ch = mt * 16 + r + 8 * hs;
      ao[((size_t)b * 64 + ch) * 4096 + qg] = acc[mt][r] * inv;
    }
}

// ---------------------------------------------------------------------------
// Kernel 4: output 1x1 conv + residual.  out = x + Wo*attn + bo
// ---------------------------------------------------------------------------
__global__ void __launch_bounds__(256)
out_proj_kernel(const float* __restrict__ x, const float* __restrict__ ao,
                const float* __restrict__ Wo, const float* __restrict__ bo,
                float* __restrict__ out) {
  int blk = blockIdx.x;
  int b = blk >> 6;
  int p0 = (blk & 63) * 64;
  __shared__ float w_s[4096];
  __shared__ float a_s[64][65];
  int tid = threadIdx.x;
  for (int i = tid; i < 4096; i += 256) {
    w_s[i] = Wo[i];
    int ch = i >> 6, pos = i & 63;
    a_s[pos][ch] = ao[((size_t)b * 64 + ch) * 4096 + p0 + pos];
  }
  __syncthreads();
  int pos = tid >> 2;
  int oc0 = (tid & 3) * 16;
  const float* ar = a_s[pos];
  for (int oc = oc0; oc < oc0 + 16; ++oc) {
    float d = bo[oc];
    const float* wr = &w_s[oc * 64];
    #pragma unroll
    for (int c = 0; c < 64; ++c) d = fmaf(ar[c], wr[c], d);
    size_t idx = ((size_t)b * 64 + oc) * 4096 + p0 + pos;
    out[idx] = x[idx] + d;
  }
}

// ---------------------------------------------------------------------------
extern "C" void kernel_launch(void* const* d_in, const int* in_sizes, int n_in,
                              void* d_out, int out_size, void* d_ws, size_t ws_size,
                              hipStream_t stream) {
  const float* x     = (const float*)d_in[0];
  const float* Wq    = (const float*)d_in[1];
  const float* bq    = (const float*)d_in[2];
  const float* Wk    = (const float*)d_in[3];
  const float* bk    = (const float*)d_in[4];
  const float* Wv    = (const float*)d_in[5];
  const float* bv    = (const float*)d_in[6];
  const float* Wo    = (const float*)d_in[7];
  const float* bo    = (const float*)d_in[8];
  const float* gamma = (const float*)d_in[9];
  const float* beta  = (const float*)d_in[10];
  float* out = (float*)d_out;

  char* ws = (char*)d_ws;
  float* stats = (float*)ws;                              // 1 KB
  h16*   q     = (h16*)(ws + 4096);                       // 2 MB
  h16*   k     = (h16*)(ws + 4096 + (1u << 21));          // 2 MB
  h16*   vT    = (h16*)(ws + 4096 + (2u << 21));          // 2 MB
  float* ao    = (float*)(ws + 4096 + (3u << 21));        // 4 MB

  gn_stats_kernel<<<128, 256, 0, stream>>>(x, stats);
  qkv_kernel<<<256, 256, 0, stream>>>(x, stats, Wq, bq, Wk, bk, Wv, bv,
                                      gamma, beta, q, k, vT);
  dim3 grid(32, 4);
  flash_kernel<<<grid, 256, 0, stream>>>(q, k, vT, ao);
  out_proj_kernel<<<256, 256, 0, stream>>>(x, ao, Wo, bo, out);
}
